// Decoder_44040594653904
// MI455X (gfx1250) — matrix operand order repaired and checked by reading, not compiled, not run on hardware
//
#include <hip/hip_runtime.h>
#include <hip/hip_bf16.h>
#include <math.h>

#define SEQ    2048
#define HD     512
#define NHEADS 6
#define VOCAB  32000
#define GRU_BLOCKS 128

#define BM 128
#define BN 256
#define BK 32
#define LDK (BK + 8)   // 40 ushorts = 80 B row stride: 16B-aligned, conflict-free frag reads

typedef __bf16  bf16_t;
typedef bf16_t        v16bf __attribute__((ext_vector_type(16)));
typedef float         v8f   __attribute__((ext_vector_type(8)));
typedef unsigned      v4u   __attribute__((ext_vector_type(4)));
typedef unsigned short v4s  __attribute__((ext_vector_type(4)));

union FragU { v4u q[2]; v16bf v; };

__device__ __forceinline__ unsigned short f2bf(float f) {
    unsigned u = __float_as_uint(f);
    u += 0x7FFFu + ((u >> 16) & 1u);           // round-to-nearest-even
    return (unsigned short)(u >> 16);
}

// ---------------------------------------------------------------------------
// Tiled GEMM: C[M,N] = alpha * A[M,K] x B (+ bias), f32 in/out, bf16 WMMA.
// Block tile 128x256x32, 8 waves as 2x4, each wave 64x64 (4x4 WMMA subtiles).
// LDS layouts give contiguous 16B fragments -> ds_load_b128, no VALU packing.
// btrans=0: B is KxN row-major. btrans=1: B is NxK row-major (A @ B^T).
// Requires M%128==0, N%256==0, K%32==0 (true for all uses here).
// ---------------------------------------------------------------------------
__global__ __launch_bounds__(256)
void gemm_bf16_wmma(const float* __restrict__ A, const float* __restrict__ B,
                    const float* __restrict__ bias, float* __restrict__ C,
                    int M, int N, int K, int lda, int ldb, int ldc,
                    int btrans, float alpha)
{
    __shared__ __align__(16) unsigned short As[BM][LDK];   // row-major, k contiguous
    __shared__ __align__(16) unsigned short Bs[BN][LDK];   // n-major,  k contiguous

    const int tid  = threadIdx.x;
    const int m0   = blockIdx.y * BM;
    const int n0   = blockIdx.x * BN;
    const int wv   = tid >> 5;        // wave 0..7
    const int lane = tid & 31;
    const int half = lane >> 4;       // wave32 lane half
    const int lr   = lane & 15;
    const int wm   = wv >> 2;         // 0..1 -> 64 rows
    const int wn   = wv & 3;          // 0..3 -> 64 cols

    const v8f vzero = {0.f,0.f,0.f,0.f,0.f,0.f,0.f,0.f};
    v8f acc[4][4];
    #pragma unroll
    for (int i = 0; i < 4; ++i)
        #pragma unroll
        for (int j = 0; j < 4; ++j) acc[i][j] = vzero;

    for (int k0 = 0; k0 < K; k0 += BK) {
        // ---- stage A tile (128 x 32), f32 -> bf16, k-contiguous ----
        #pragma unroll
        for (int it = 0; it < 4; ++it) {
            int s  = tid + it * 256;          // 1024 float4 slots
            int ar = s >> 3;                  // 0..127
            int ac = (s & 7) * 4;             // 0..28
            const float4 g = *(const float4*)(A + (size_t)(m0 + ar) * lda + k0 + ac);
            v4s t = { f2bf(g.x), f2bf(g.y), f2bf(g.z), f2bf(g.w) };
            *(v4s*)&As[ar][ac] = t;
        }
        // ---- stage B tile as Bs[n][k] (n-major) ----
        if (btrans) {
            // B is NxK row-major: k runs contiguous in global -> direct vec4 copy
            #pragma unroll
            for (int it = 0; it < 8; ++it) {
                int s  = tid + it * 256;      // 2048 float4 slots
                int nr = s >> 3;              // 0..255
                int kc = (s & 7) * 4;         // 0..28
                const float4 g = *(const float4*)(B + (size_t)(n0 + nr) * ldb + k0 + kc);
                v4s t = { f2bf(g.x), f2bf(g.y), f2bf(g.z), f2bf(g.w) };
                *(v4s*)&Bs[nr][kc] = t;
            }
        } else {
            // B is KxN row-major: read along n (coalesced across lanes), scatter-transpose
            #pragma unroll
            for (int it = 0; it < 8; ++it) {
                int s  = tid + it * 256;      // 2048 float4 slots
                int kr = s >> 6;              // 0..31
                int nq = (s & 63) * 4;        // 0..252
                const float4 g = *(const float4*)(B + (size_t)(k0 + kr) * ldb + n0 + nq);
                Bs[nq + 0][kr] = f2bf(g.x);
                Bs[nq + 1][kr] = f2bf(g.y);
                Bs[nq + 2][kr] = f2bf(g.z);
                Bs[nq + 3][kr] = f2bf(g.w);
            }
        }
        // prefetch next K tile (global_prefetch_b8)
        if (k0 + BK < K) {
            __builtin_prefetch(A + (size_t)(m0 + (tid >> 1)) * lda + k0 + BK, 0, 1);
            if (btrans)
                __builtin_prefetch(B + (size_t)(n0 + tid) * ldb + k0 + BK, 0, 1);
            else
                __builtin_prefetch(B + (size_t)(k0 + BK + (tid >> 3)) * ldb + n0 + ((tid & 7) << 5), 0, 1);
        }
        __syncthreads();

        // ---- B fragments: 2x ds_load_b128 each (K = half*16 .. half*16+15) ----
        FragU bfr[4];
        #pragma unroll
        for (int ni = 0; ni < 4; ++ni) {
            const int n = wn * 64 + ni * 16 + lr;
            const v4u* pb = (const v4u*)&Bs[n][0];
            bfr[ni].q[0] = pb[2 * half];
            bfr[ni].q[1] = pb[2 * half + 1];
        }
        // ---- A fragments (2x b128: K = half*16 .. half*16+15) + WMMA ----
        #pragma unroll
        for (int mi = 0; mi < 4; ++mi) {
            const int row = wm * 64 + mi * 16 + lr;
            const v4u* pa = (const v4u*)&As[row][0];
            FragU af;
            af.q[0] = pa[2 * half];        // bytes [half*32, half*32+16)
            af.q[1] = pa[2 * half + 1];    // bytes [half*32+16, half*32+32)
            #pragma unroll
            for (int ni = 0; ni < 4; ++ni) {
                acc[mi][ni] = __builtin_amdgcn_wmma_f32_16x16x32_bf16(
                    false, af.v, false, bfr[ni].v, (short)0, acc[mi][ni], false, false);
            }
        }
        __syncthreads();
    }

    // ---- epilogue: C layout VGPR r -> M = r + 8*half, N = lane&15 ----
    #pragma unroll
    for (int mi = 0; mi < 4; ++mi) {
        #pragma unroll
        for (int ni = 0; ni < 4; ++ni) {
            const int col = n0 + wn * 64 + ni * 16 + lr;
            const float bv = bias ? bias[col] : 0.0f;
            #pragma unroll
            for (int r = 0; r < 8; ++r) {
                const int row = m0 + wm * 64 + mi * 16 + half * 8 + r;
                C[(size_t)row * ldc + col] = acc[mi][ni][r] * alpha + bv;
            }
        }
    }
}

// ---------------------------------------------------------------------------
__global__ __launch_bounds__(256)
void embed_kernel(const int* __restrict__ tok, const float* __restrict__ emb,
                  float* __restrict__ x)
{
    const int s = blockIdx.x;
    const int t = tok[s];
    for (int i = threadIdx.x; i < HD; i += 256)
        x[(size_t)s * HD + i] = emb[(size_t)t * HD + i];
}

// block per row, softmax in place over n columns
__global__ __launch_bounds__(256)
void softmax_rows(float* __restrict__ p, int n)
{
    float* row = p + (size_t)blockIdx.x * n;
    __shared__ float red[256];
    __shared__ float stat;
    const int tid = threadIdx.x;

    float m = -3.0e38f;
    for (int i = tid; i < n; i += 256) m = fmaxf(m, row[i]);
    red[tid] = m; __syncthreads();
    for (int s = 128; s > 0; s >>= 1) { if (tid < s) red[tid] = fmaxf(red[tid], red[tid + s]); __syncthreads(); }
    if (tid == 0) stat = red[0];
    __syncthreads();
    const float mx = stat;

    float sum = 0.f;
    for (int i = tid; i < n; i += 256) { float e = __expf(row[i] - mx); row[i] = e; sum += e; }
    __syncthreads();
    red[tid] = sum; __syncthreads();
    for (int s = 128; s > 0; s >>= 1) { if (tid < s) red[tid] += red[tid + s]; __syncthreads(); }
    if (tid == 0) stat = 1.0f / red[0];
    __syncthreads();
    const float inv = stat;
    for (int i = tid; i < n; i += 256) row[i] *= inv;
}

// y = x + mha; layernorm(y) -> out   (block per row, H=512, 2 elems/thread)
__global__ __launch_bounds__(256)
void add_ln_kernel(const float* __restrict__ x, const float* __restrict__ mha,
                   const float* __restrict__ gamma, const float* __restrict__ beta,
                   float* __restrict__ out)
{
    const int s   = blockIdx.x;
    const int tid = threadIdx.x;
    __shared__ float red[256];
    __shared__ float stat;

    const size_t base = (size_t)s * HD;
    float y0 = x[base + tid]       + mha[base + tid];
    float y1 = x[base + 256 + tid] + mha[base + 256 + tid];

    red[tid] = y0 + y1; __syncthreads();
    for (int st = 128; st > 0; st >>= 1) { if (tid < st) red[tid] += red[tid + st]; __syncthreads(); }
    if (tid == 0) stat = red[0] * (1.0f / HD);
    __syncthreads();
    const float mu = stat;

    float d0 = y0 - mu, d1 = y1 - mu;
    red[tid] = d0 * d0 + d1 * d1; __syncthreads();
    for (int st = 128; st > 0; st >>= 1) { if (tid < st) red[tid] += red[tid + st]; __syncthreads(); }
    if (tid == 0) stat = rsqrtf(red[0] * (1.0f / HD) + 1e-5f);
    __syncthreads();
    const float rstd = stat;

    out[base + tid]       = d0 * rstd * gamma[tid]       + beta[tid];
    out[base + 256 + tid] = d1 * rstd * gamma[256 + tid] + beta[256 + tid];
}

// ---------------------------------------------------------------------------
__global__ __launch_bounds__(512)
void gru_init_kernel(const float* __restrict__ dec_hid, float* __restrict__ hb0,
                     int* __restrict__ counter)
{
    const int i = threadIdx.x;
    hb0[i] = dec_hid[i];
    if (i == 0) *counter = 0;
}

// Persistent lockstep GRU: 128 blocks x 384 threads (12 waves).
// Block b owns output channels e = 4b..4b+3; caches its 12 W_hh rows
// (3 gates x 4 channels, 24 KB) in LDS once; per step: one wave per row
// does a lane-strided dot + wave32 shuffle reduce; grid barrier via atomic.
__global__ __launch_bounds__(384)
void gru_kernel(const float* __restrict__ gi_all, const float* __restrict__ W_hh,
                const float* __restrict__ b_hh,
                float* __restrict__ hb0, float* __restrict__ hb1,
                int* __restrict__ counter, float* __restrict__ h_last)
{
    __shared__ float Ws[12][HD];
    __shared__ float hs[HD];
    __shared__ float ghs[12];

    const int tid  = threadIdx.x;
    const int wv   = tid >> 5;     // 0..11
    const int lane = tid & 31;
    const int eb   = blockIdx.x * 4;

    // cache W_hh slice: row j = gate*4 + e_local  ->  W_hh[gate*512 + eb + e_local]
    for (int idx = tid; idx < 12 * HD; idx += 384) {
        int j = idx >> 9, k = idx & (HD - 1);
        int gate = j >> 2, el = j & 3;
        Ws[j][k] = W_hh[(size_t)(gate * HD + eb + el) * HD + k];
    }
    __syncthreads();

    float* bufs[2] = { hb0, hb1 };
    for (int t = 0; t < SEQ; ++t) {
        const float* hsrc = bufs[t & 1];
        float*       hdst = bufs[(t + 1) & 1];

        for (int i = tid; i < HD; i += 384) hs[i] = hsrc[i];
        __syncthreads();

        // wave wv: dot(Ws[wv], h)
        float s = 0.f;
        const float* wrow = Ws[wv];
        #pragma unroll
        for (int k = lane; k < HD; k += 32) s += wrow[k] * hs[k];
        #pragma unroll
        for (int off = 16; off > 0; off >>= 1) s += __shfl_down(s, off, 32);
        if (lane == 0) ghs[wv] = s;
        __syncthreads();

        if (tid < 4) {
            const int e = eb + tid;
            const size_t gb = (size_t)t * (3 * HD);
            float gir = gi_all[gb + e];
            float giz = gi_all[gb + HD + e];
            float gin = gi_all[gb + 2 * HD + e];
            float ghr = ghs[0 + tid] + b_hh[e];
            float ghz = ghs[4 + tid] + b_hh[HD + e];
            float ghn = ghs[8 + tid] + b_hh[2 * HD + e];
            float r = 1.0f / (1.0f + __expf(-(gir + ghr)));
            float z = 1.0f / (1.0f + __expf(-(giz + ghz)));
            float n = tanhf(gin + r * ghn);
            hdst[e] = (1.0f - z) * n + z * hs[e];
            __threadfence();
        }
        __syncthreads();

        // grid barrier (lockstep step t)
        if (tid == 0) {
            atomicAdd(counter, 1);
            const int target = GRU_BLOCKS * (t + 1);
            volatile int* vc = counter;
            while (*vc < target) { __builtin_amdgcn_s_sleep(2); }
            __threadfence();
        }
        __syncthreads();
    }

    if (blockIdx.x == 0)
        for (int i = tid; i < HD; i += 384) h_last[i] = hb0[i];   // SEQ even -> final in hb0
}

// ---------------------------------------------------------------------------
__global__ __launch_bounds__(256)
void logits_kernel(const float* __restrict__ h, const float* __restrict__ W_out,
                   const float* __restrict__ b_out, float* __restrict__ logits)
{
    __shared__ float hs[HD];
    const int tid = threadIdx.x;
    for (int i = tid; i < HD; i += 256) hs[i] = h[i];
    __syncthreads();
    const int v = blockIdx.x * 256 + tid;
    float s = b_out[v];
    #pragma unroll 4
    for (int d = 0; d < HD; ++d) s += hs[d] * W_out[(size_t)d * VOCAB + v];
    logits[v] = s;
}

__global__ __launch_bounds__(1024)
void logsoftmax_kernel(const float* __restrict__ logits, const float* __restrict__ h,
                       float* __restrict__ out)
{
    __shared__ float red[1024];
    __shared__ float stat;
    const int tid = threadIdx.x;

    float m = -3.0e38f;
    for (int v = tid; v < VOCAB; v += 1024) m = fmaxf(m, logits[v]);
    red[tid] = m; __syncthreads();
    for (int s = 512; s > 0; s >>= 1) { if (tid < s) red[tid] = fmaxf(red[tid], red[tid + s]); __syncthreads(); }
    if (tid == 0) stat = red[0];
    __syncthreads();
    const float mx = stat;

    float sum = 0.f;
    for (int v = tid; v < VOCAB; v += 1024) sum += __expf(logits[v] - mx);
    __syncthreads();
    red[tid] = sum; __syncthreads();
    for (int s = 512; s > 0; s >>= 1) { if (tid < s) red[tid] += red[tid + s]; __syncthreads(); }
    if (tid == 0) stat = mx + __logf(red[0]);
    __syncthreads();
    const float lse = stat;

    for (int v = tid; v < VOCAB; v += 1024) out[v] = logits[v] - lse;
    for (int i = tid; i < HD; i += 1024) out[VOCAB + i] = h[i];
}

// ---------------------------------------------------------------------------
extern "C" void kernel_launch(void* const* d_in, const int* in_sizes, int n_in,
                              void* d_out, int out_size, void* d_ws, size_t ws_size,
                              hipStream_t stream)
{
    (void)in_sizes; (void)n_in; (void)out_size; (void)ws_size;
    const int*   tok     = (const int*)  d_in[0];
    const float* dec_hid = (const float*)d_in[1];
    const float* emb     = (const float*)d_in[2];
    const float* Wq      = (const float*)d_in[3];
    const float* bq      = (const float*)d_in[4];
    const float* Wk      = (const float*)d_in[5];
    const float* bk      = (const float*)d_in[6];
    const float* Wv      = (const float*)d_in[7];
    const float* bv      = (const float*)d_in[8];
    const float* Wo      = (const float*)d_in[9];
    const float* bo      = (const float*)d_in[10];
    const float* gamma   = (const float*)d_in[11];
    const float* beta    = (const float*)d_in[12];
    const float* W_ih    = (const float*)d_in[13];
    const float* W_hh    = (const float*)d_in[14];
    const float* b_ih    = (const float*)d_in[15];
    const float* b_hh    = (const float*)d_in[16];
    const float* W_out   = (const float*)d_in[17];
    const float* b_out   = (const float*)d_in[18];
    float* out = (float*)d_out;

    float* ws = (float*)d_ws;
    size_t off = 0;
    float* x     = ws + off; off += (size_t)SEQ * HD;
    float* xln   = ws + off; off += (size_t)SEQ * HD;
    float* q     = ws + off; off += (size_t)SEQ * HD;
    float* kbuf  = ws + off; off += (size_t)SEQ * HD;
    float* vbuf  = ws + off; off += (size_t)SEQ * HD;
    float* sc    = ws + off; off += (size_t)SEQ * SEQ;
    float* cat   = ws + off; off += (size_t)SEQ * NHEADS * HD;
    float* mha   = ws + off; off += (size_t)SEQ * HD;
    float* gi    = ws + off; off += (size_t)SEQ * 3 * HD;
    float* hb0   = ws + off; off += HD;
    float* hb1   = ws + off; off += HD;
    float* hlast = ws + off; off += HD;
    float* lg    = ws + off; off += VOCAB;
    int*   counter = (int*)(ws + off);

    embed_kernel<<<SEQ, 256, 0, stream>>>(tok, emb, x);

    const float scl = 1.0f / sqrtf((float)HD);
    const dim3 gNH(HD / BN, SEQ / BM);    // N=512 tiles (2 x 16)
    const dim3 gNS(SEQ / BN, SEQ / BM);   // N=2048 tiles (8 x 16)

    for (int h = 0; h < NHEADS; ++h) {
        gemm_bf16_wmma<<<gNH, 256, 0, stream>>>(x, Wq + (size_t)h * HD * HD, bq + h * HD,
                                                q,    SEQ, HD, HD, HD, HD, HD, 0, 1.0f);
        gemm_bf16_wmma<<<gNH, 256, 0, stream>>>(x, Wk + (size_t)h * HD * HD, bk + h * HD,
                                                kbuf, SEQ, HD, HD, HD, HD, HD, 0, 1.0f);
        gemm_bf16_wmma<<<gNH, 256, 0, stream>>>(x, Wv + (size_t)h * HD * HD, bv + h * HD,
                                                vbuf, SEQ, HD, HD, HD, HD, HD, 0, 1.0f);
        // scores = q @ k^T * scale   (B transposed)
        gemm_bf16_wmma<<<gNS, 256, 0, stream>>>(q, kbuf, nullptr, sc,
                                                SEQ, SEQ, HD, HD, HD, SEQ, 1, scl);
        softmax_rows<<<SEQ, 256, 0, stream>>>(sc, SEQ);
        // heads = p @ v  -> written directly into cat layout (ldc = NH*H)
        gemm_bf16_wmma<<<gNH, 256, 0, stream>>>(sc, vbuf, nullptr, cat + h * HD,
                                                SEQ, HD, SEQ, SEQ, HD, NHEADS * HD, 0, 1.0f);
    }

    gemm_bf16_wmma<<<gNH, 256, 0, stream>>>(cat, Wo, bo, mha,
                                            SEQ, HD, NHEADS * HD, NHEADS * HD, HD, HD, 0, 1.0f);
    add_ln_kernel<<<SEQ, 256, 0, stream>>>(x, mha, gamma, beta, xln);

    // gi_all = xln @ W_ih^T + b_ih   (B transposed, N=1536)
    gemm_bf16_wmma<<<dim3(3 * HD / BN, SEQ / BM), 256, 0, stream>>>(
        xln, W_ih, b_ih, gi, SEQ, 3 * HD, HD, HD, HD, 3 * HD, 1, 1.0f);

    gru_init_kernel<<<1, 512, 0, stream>>>(dec_hid, hb0, counter);
    gru_kernel<<<GRU_BLOCKS, 384, 0, stream>>>(gi, W_hh, b_hh, hb0, hb1, counter, hlast);

    logits_kernel<<<VOCAB / 256, 256, 0, stream>>>(hlast, W_out, b_out, lg);
    logsoftmax_kernel<<<1, 1024, 0, stream>>>(lg, hlast, out);
}
